// JKNet_75797582840808
// MI455X (gfx1250) — matrix operand order repaired and checked by reading, not compiled, not run on hardware
//
#include <hip/hip_runtime.h>
#include <hip/hip_bf16.h>
#include <math.h>

#define NNODES 100000
#define NEDGES 800000
#define DIM    128
#define NLAYER 5
#define NCLS   40
#define NCLSP  48        // padded to 3 WMMA column tiles
#define BN_EPS 1e-5f

typedef __attribute__((ext_vector_type(16))) _Float16 v16h;
typedef __attribute__((ext_vector_type(8)))  _Float16 v8h;
typedef __attribute__((ext_vector_type(8)))  float    v8f;

// ---------------------------------------------------------------------------
// WMMA fragment loaders (ISA 7.12.2 layouts, wave32)
// A (16x32 f16): lanes 0-15 -> row M=lane, K = {0..7, 16..23}+k0
//                lanes 16-31 -> row M=lane-16, K = {8..15, 24..31}+k0
__device__ __forceinline__ v16h load_a_frag(const _Float16* As, int lane, int k0) {
    int r  = lane & 15;
    int hs = lane >> 4;
    const _Float16* p = As + r * DIM + k0 + hs * 8;
    v8h lo = *(const v8h*)(p);        // K = base .. base+7
    v8h hi = *(const v8h*)(p + 16);   // K = base+16 .. base+23
    return __builtin_shufflevector(lo, hi, 0,1,2,3,4,5,6,7,8,9,10,11,12,13,14,15);
}

// B (32x16 f16), Wt stored transposed [col][K]: lanes 0-15 -> K = {0..7, 16..23}+k0,
// lanes 16-31 -> K = {8..15, 24..31}+k0; column N = lane&15 (+tile base)
__device__ __forceinline__ v16h load_b_frag(const _Float16* Wt, int lane, int colbase, int k0) {
    int col = colbase + (lane & 15);
    int hs  = lane >> 4;
    const _Float16* p = Wt + col * DIM + k0 + hs * 8;
    v8h lo = *(const v8h*)(p);
    v8h hi = *(const v8h*)(p + 16);
    return __builtin_shufflevector(lo, hi, 0,1,2,3,4,5,6,7,8,9,10,11,12,13,14,15);
}

// ---------------------------------------------------------------------------
// Tiled GEMM: out[rowbase+0..15][0..NC) = A[16xK=128] @ W[128 x WC]
// One wave per 16-col tile, f16 inputs, f32 WMMA accumulation.
template<int NC, int WC, int NT>
__global__ void jk_gemm_wmma(const float* __restrict__ A,
                             const float* __restrict__ W,
                             float* __restrict__ out, int ldo) {
    __shared__ __align__(32) _Float16 As[16 * DIM];
    __shared__ __align__(32) _Float16 Wt[NC * DIM];

    const int tid     = threadIdx.x;
    const int rowbase = blockIdx.x * 16;

    // stage A tile (coalesced, f32 -> f16)
    for (int i = tid; i < 16 * DIM; i += NT) {
        int r = i >> 7, c = i & (DIM - 1);
        As[i] = (_Float16)A[(rowbase + r) * DIM + c];
    }
    // stage weights transposed: Wt[c][k] = W[k][c]; pad cols >= WC with 0
    for (int i = tid; i < NC * DIM; i += NT) {
        int c = i >> 7, k = i & (DIM - 1);
        Wt[i] = (c < WC) ? (_Float16)W[k * WC + c] : (_Float16)0.0f;
    }
    __syncthreads();

    const int wave    = tid >> 5;
    const int lane    = tid & 31;
    const int colbase = wave * 16;

    v8f acc = {};
#pragma unroll
    for (int k0 = 0; k0 < DIM; k0 += 32) {
        v16h a = load_a_frag(As, lane, k0);
        v16h b = load_b_frag(Wt, lane, colbase, k0);
        acc = __builtin_amdgcn_wmma_f32_16x16x32_f16(
                false, a, false, b, (short)0, acc, false, false);
    }

    // D layout: VGPR v -> row (v | hs*8), col = colbase + (lane&15)
    const int n  = colbase + (lane & 15);
    const int hs = lane >> 4;
#pragma unroll
    for (int v = 0; v < 8; ++v) {
        int m = v + hs * 8;
        out[(rowbase + m) * ldo + n] = acc[v];
    }
}

// ---------------------------------------------------------------------------
__global__ void jk_zero(float* __restrict__ p, int n) {
    int i = blockIdx.x * blockDim.x + threadIdx.x;
    if (i < n) p[i] = 0.0f;
}

__global__ void jk_count_deg(const int* __restrict__ ei, float* __restrict__ deg) {
    int e = blockIdx.x * blockDim.x + threadIdx.x;
    if (e < NEDGES) atomicAdd(&deg[ei[e]], 1.0f);   // ei[0..E) = row (dst)
}

__global__ void jk_dinv(float* __restrict__ deg, int n) {
    int i = blockIdx.x * blockDim.x + threadIdx.x;
    if (i < n) deg[i] = rsqrtf(deg[i] + 1.0f);      // in-place -> dinv
}

// agg = hw * dinv^2 (self loop) + bias     [runs before edge atomics]
__global__ void jk_agg_init(const float* __restrict__ hw,
                            const float* __restrict__ dinv,
                            const float* __restrict__ bias,
                            float* __restrict__ agg) {
    int idx = blockIdx.x * blockDim.x + threadIdx.x;   // exact: N*D/256 blocks
    int n = idx >> 7, d = idx & (DIM - 1);
    float dv = dinv[n];
    agg[idx] = hw[idx] * dv * dv + bias[d];
}

// one wave32 per edge: gather hw[col] (float4/lane) and atomically add into agg[row]
__global__ void jk_edge_agg(const int* __restrict__ ei,
                            const float* __restrict__ dinv,
                            const float* __restrict__ hw,
                            float* __restrict__ agg) {
    int gtid = blockIdx.x * blockDim.x + threadIdx.x;
    int e    = gtid >> 5;
    if (e >= NEDGES) return;
    int lane = gtid & 31;
    int r = ei[e];            // dst
    int c = ei[NEDGES + e];   // src
    float w = dinv[r] * dinv[c];
    const float4 v = *(const float4*)&hw[c * DIM + lane * 4];
    float* dst = &agg[r * DIM + lane * 4];
    atomicAdd(dst + 0, v.x * w);
    atomicAdd(dst + 1, v.y * w);
    atomicAdd(dst + 2, v.z * w);
    atomicAdd(dst + 3, v.w * w);
}

// BN(eval) + ReLU, write next-layer h, fold into JumpingKnowledge max
__global__ void jk_bn_relu_max(const float* __restrict__ agg,
                               const float* __restrict__ gamma,
                               const float* __restrict__ beta,
                               const float* __restrict__ mean,
                               const float* __restrict__ var,
                               float* __restrict__ h,
                               float* __restrict__ jk,
                               int first_layer) {
    int idx = blockIdx.x * blockDim.x + threadIdx.x;   // exact grid
    int d = idx & (DIM - 1);
    float v = gamma[d] * (agg[idx] - mean[d]) * rsqrtf(var[d] + BN_EPS) + beta[d];
    v = fmaxf(v, 0.0f);
    h[idx] = v;
    jk[idx] = first_layer ? v : fmaxf(jk[idx], v);
}

// per-node log_softmax over NCLS classes (logits padded to NCLSP)
__global__ void jk_logsoftmax(const float* __restrict__ logits,
                              const float* __restrict__ lin_b,
                              float* __restrict__ out) {
    int n = blockIdx.x * blockDim.x + threadIdx.x;
    if (n >= NNODES) return;
    float z[NCLS];
    float mx = -INFINITY;
#pragma unroll
    for (int c = 0; c < NCLS; ++c) {
        z[c] = logits[n * NCLSP + c] + lin_b[c];
        mx = fmaxf(mx, z[c]);
    }
    float s = 0.0f;
#pragma unroll
    for (int c = 0; c < NCLS; ++c) s += expf(z[c] - mx);
    float lse = mx + logf(s);
#pragma unroll
    for (int c = 0; c < NCLS; ++c) out[n * NCLS + c] = z[c] - lse;
}

// ---------------------------------------------------------------------------
extern "C" void kernel_launch(void* const* d_in, const int* in_sizes, int n_in,
                              void* d_out, int out_size, void* d_ws, size_t ws_size,
                              hipStream_t stream) {
    const float* x      = (const float*)d_in[0];
    const int*   ei     = (const int*)  d_in[1];   // [2,E]: row then col
    const float* conv_w = (const float*)d_in[2];   // [L,D,D]
    const float* conv_b = (const float*)d_in[3];   // [L,D]
    const float* gamma  = (const float*)d_in[4];
    const float* beta   = (const float*)d_in[5];
    const float* mean   = (const float*)d_in[6];
    const float* var    = (const float*)d_in[7];
    const float* lin_w  = (const float*)d_in[8];   // [D,C]
    const float* lin_b  = (const float*)d_in[9];   // [C]
    float*       out    = (float*)d_out;

    // workspace layout (floats)
    float* ws     = (float*)d_ws;
    float* dinv   = ws;                       // N      (deg -> dinv in place)
    float* h      = dinv + NNODES;            // N*D
    float* hw     = h    + (size_t)NNODES * DIM;
    float* agg    = hw   + (size_t)NNODES * DIM;
    float* jk     = agg  + (size_t)NNODES * DIM;
    float* logits = jk   + (size_t)NNODES * DIM;  // N * NCLSP

    const int ND        = NNODES * DIM;            // 12.8M, exact /256
    const int rowTiles  = NNODES / 16;             // 6250
    const int edgeBlks  = NEDGES / 8;              // 8 waves (edges) per block

    // degrees -> dinv
    jk_zero<<<(NNODES + 255) / 256, 256, 0, stream>>>(dinv, NNODES);
    jk_count_deg<<<NEDGES / 256, 256, 0, stream>>>(ei, dinv);
    jk_dinv<<<(NNODES + 255) / 256, 256, 0, stream>>>(dinv, NNODES);

    // 5 GCN layers
    for (int l = 0; l < NLAYER; ++l) {
        const float* hin = (l == 0) ? x : h;
        jk_gemm_wmma<DIM, DIM, 256><<<rowTiles, 256, 0, stream>>>(
            hin, conv_w + (size_t)l * DIM * DIM, hw, DIM);
        jk_agg_init<<<ND / 256, 256, 0, stream>>>(hw, dinv, conv_b + l * DIM, agg);
        jk_edge_agg<<<edgeBlks, 256, 0, stream>>>(ei, dinv, hw, agg);
        jk_bn_relu_max<<<ND / 256, 256, 0, stream>>>(
            agg, gamma + l * DIM, beta + l * DIM, mean + l * DIM, var + l * DIM,
            h, jk, l == 0 ? 1 : 0);
    }

    // head: jk[N,128] @ lin_w[128,40] -> padded logits, then log_softmax
    jk_gemm_wmma<NCLSP, NCLS, 96><<<rowTiles, 96, 0, stream>>>(jk, lin_w, logits, NCLSP);
    jk_logsoftmax<<<(NNODES + 255) / 256, 256, 0, stream>>>(logits, lin_b, out);
}
